// DenoiseNet_79817672229589
// MI455X (gfx1250) — matrix operand order repaired and checked by reading, not compiled, not run on hardware
//
#include <hip/hip_runtime.h>
#include <hip/hip_bf16.h>
#include <math.h>

typedef __attribute__((ext_vector_type(16))) _Float16 v16h;
typedef __attribute__((ext_vector_type(8)))  _Float16 v8h;
typedef __attribute__((ext_vector_type(8)))  float    v8f;
typedef int v4i __attribute__((vector_size(16)));

#define NPTS   32000L   // 32 batches * 1000 points
#define BATCH  32
#define PPB    1000
#define KC     128      // K-chunk staged in LDS
#define KCP    136      // padded LDS row stride (halves): 4-bank skew per column

// ---------------------------------------------------------------------------
// CDNA5 async global->LDS staging (ASYNCcnt-tracked), with safe fallback.
// Builtin signature (from clang diagnostic): (global v4i*, lds v4i*, Ii, Ii)
// ---------------------------------------------------------------------------
#if defined(__HIP_DEVICE_COMPILE__) && __has_builtin(__builtin_amdgcn_global_load_async_to_lds_b128)
#define HAS_ASYNC_LDS 1
#else
#define HAS_ASYNC_LDS 0
#endif

__device__ inline void stage16(const _Float16* g, _Float16* l) {
#if HAS_ASYNC_LDS
    __builtin_amdgcn_global_load_async_to_lds_b128(
        (__attribute__((address_space(1))) v4i*)g,
        (__attribute__((address_space(3))) v4i*)l, 0, 0);
#else
    *(v8h*)l = *(const v8h*)g;
#endif
}
__device__ inline void stage_wait() {
#if HAS_ASYNC_LDS
#if __has_builtin(__builtin_amdgcn_s_wait_asynccnt)
    __builtin_amdgcn_s_wait_asynccnt(0);
#else
    asm volatile("s_wait_asynccnt 0x0" ::: "memory");
#endif
#endif
}

// ---------------------------------------------------------------------------
// Fragment loaders (wave32 WMMA ISA layouts).
// A 16x32 f16: per lane two 8-half segments 16 halves apart (K and K+16).
// B 32x16 f16: per lane two 8-half segments 16 halves apart (K and K+16).
// ---------------------------------------------------------------------------
__device__ inline v16h ldA(const _Float16* p) {
    union { v16h v; v8h h[2]; } u;
    u.h[0] = *(const v8h*)p;
    u.h[1] = *(const v8h*)(p + 16);
    return u.v;
}
__device__ inline v16h ldB(const _Float16* p) {
    union { v16h v; v8h h[2]; } u;
    u.h[0] = *(const v8h*)p;
    u.h[1] = *(const v8h*)(p + 16);
    return u.v;
}

// ---------------------------------------------------------------------------
__global__ void zero_out_k(float* out) { out[0] = 0.0f; }

// ---------------------------------------------------------------------------
// Weight prep: f32 (fi x fo) -> f16 transposed (N rows x K cols), zero padded.
// blockIdx.z = model iteration (M=4 stacked weight sets).
// ---------------------------------------------------------------------------
__global__ __launch_bounds__(256)
void prep_wt_k(const float* __restrict__ w, _Float16* __restrict__ wt,
               int fi, int fo, int K, int N, long iterStrideHalves)
{
    long idx = (long)blockIdx.x * 256 + threadIdx.x;
    long tot = (long)N * K;
    if (idx >= tot) return;
    int n = (int)(idx / K);
    int k = (int)(idx % K);
    int z = blockIdx.z;
    float v = (k < fi && n < fo) ? w[(long)z * fi * fo + (long)k * fo + n] : 0.0f;
    wt[(long)z * iterStrideHalves + idx] = (_Float16)v;
}

// ---------------------------------------------------------------------------
// Seed weights sw = normalized exp(-9*d2/d2_last) per batch
// ---------------------------------------------------------------------------
__global__ __launch_bounds__(256)
void seed_sw_k(const float* __restrict__ noisy, const float* __restrict__ seeds,
               float* __restrict__ sw)
{
    __shared__ float red[256];
    __shared__ float maxsq;
    __shared__ float ssum;
    int b = blockIdx.x, tid = threadIdx.x;
    float sx = seeds[b*3+0], sy = seeds[b*3+1], sz = seeds[b*3+2];
    if (tid == 0) {
        int base = (b*PPB + PPB - 1) * 3;
        float dx = noisy[base+0]-sx, dy = noisy[base+1]-sy, dz = noisy[base+2]-sz;
        maxsq = dx*dx + dy*dy + dz*dz;
    }
    __syncthreads();
    float inv = -9.0f / maxsq;
    float acc = 0.0f;
    for (int q = tid; q < PPB; q += 256) {
        int base = (b*PPB + q)*3;
        float dx = noisy[base+0]-sx, dy = noisy[base+1]-sy, dz = noisy[base+2]-sz;
        float e = expf((dx*dx + dy*dy + dz*dz) * inv);
        sw[b*PPB + q] = e;
        acc += e;
    }
    red[tid] = acc; __syncthreads();
    for (int s = 128; s > 0; s >>= 1) { if (tid < s) red[tid] += red[tid+s]; __syncthreads(); }
    if (tid == 0) ssum = red[0];
    __syncthreads();
    float r = 1.0f / ssum;
    for (int q = tid; q < PPB; q += 256) sw[b*PPB + q] *= r;
}

// ---------------------------------------------------------------------------
// Iteration input: iter0: pin = noisy - seed; else pin += pred.
// Emits padded (K=32) f16 A-matrix for encoder layer 0.
// ---------------------------------------------------------------------------
__global__ __launch_bounds__(256)
void prep_input_k(const float* __restrict__ noisy, const float* __restrict__ seeds,
                  const float* __restrict__ pred, float* __restrict__ pin,
                  _Float16* __restrict__ act0, int iter)
{
    int idx = blockIdx.x * 256 + threadIdx.x;
    if (idx >= (int)NPTS) return;
    int b = idx / PPB;
    float p[3];
#pragma unroll
    for (int d = 0; d < 3; ++d) {
        float v = (iter == 0) ? (noisy[idx*3+d] - seeds[b*3+d])
                              : (pin[idx*3+d] + pred[idx*3+d]);
        p[d] = v;
    }
#pragma unroll
    for (int d = 0; d < 3; ++d) pin[idx*3+d] = p[d];
    _Float16* ap = act0 + (size_t)idx * 32;
#pragma unroll
    for (int d = 0; d < 3; ++d)  ap[d] = (_Float16)p[d];
#pragma unroll
    for (int d = 3; d < 32; ++d) ap[d] = (_Float16)0.0f;
}

// ---------------------------------------------------------------------------
// Main WMMA GEMM (N >= 64): Y(f32) = A(f16) * Wt^T + bias.
// Block = 8 waves sharing a 64-column B panel, double-buffered in LDS and
// staged with async global->LDS loads (copy/compute overlap).  Each wave:
// 2 M-tiles x 4 N-tiles = 8 v_wmma_f32_16x16x32_f16 per K-step, branch-free.
// ---------------------------------------------------------------------------
__global__ __launch_bounds__(256)
void gemm_lds_k(const _Float16* __restrict__ A, const _Float16* __restrict__ Wt,
                const float* __restrict__ bias, float* __restrict__ Y,
                int K, int N, int Nq)
{
    __shared__ _Float16 lbs[2][64 * KCP];   // 2 x 17 KB B panels
    const int tid  = threadIdx.x;
    const int lane = tid & 31;
    const int wv   = tid >> 5;
    const int mg   = blockIdx.x / Nq;
    const int nq   = blockIdx.x % Nq;
    const int nBase = nq * 64;
    const int m0    = (mg * 16 + wv * 2) * 16;   // wave's first row (2 M-tiles)

    const int aSel = (lane & 16) ? 8 : 0;
    const int bOff = (lane & 16) ? 8 : 0;
    const int nCol = lane & 15;
    const _Float16* aPtr0 = A + (size_t)(m0 + nCol) * K + aSel;
    const _Float16* aPtr1 = aPtr0 + (size_t)16 * K;
    const _Float16* wPanel = Wt + (size_t)nBase * K;

    const int nChunks = (K + KC - 1) / KC;
    auto stageChunk = [&](int ci) {
        int k0 = ci * KC;
        int c = K - k0; if (c > KC) c = KC;    // chunk length (mult of 32)
        int upc = c >> 3;                      // 16B units per column
        int units = upc << 6;                  // 64 * upc
        _Float16* dst = lbs[ci & 1];
        for (int u = tid; u < units; u += 256) {
            int col = u / upc;
            int off = (u - col * upc) << 3;
            stage16(wPanel + (size_t)col * K + k0 + off, dst + col * KCP + off);
        }
    };

    v8f zero = {0.f,0.f,0.f,0.f,0.f,0.f,0.f,0.f};
    v8f acc[2][4];
#pragma unroll
    for (int m = 0; m < 2; ++m)
#pragma unroll
        for (int t = 0; t < 4; ++t) acc[m][t] = zero;

    stageChunk(0);
    stage_wait();
    __syncthreads();

    for (int ci = 0; ci < nChunks; ++ci) {
        int k0 = ci * KC;
        int c = K - k0; if (c > KC) c = KC;
        if (ci + 1 < nChunks) {
            stageChunk(ci + 1);                       // overlap with compute
            __builtin_prefetch(aPtr0 + k0 + KC, 0, 1);
        }
        const _Float16* lb = lbs[ci & 1];
        for (int kk = 0; kk < c; kk += 32) {
            v16h a0 = ldA(aPtr0 + k0 + kk);
            v16h a1 = ldA(aPtr1 + k0 + kk);
#pragma unroll
            for (int t = 0; t < 4; ++t) {
                const _Float16* bp = lb + (nCol + 16*t) * KCP + kk + bOff;
                v16h b = ldB(bp);
                acc[0][t] = __builtin_amdgcn_wmma_f32_16x16x32_f16(
                                false, a0, false, b, (short)0, acc[0][t], false, false);
                acc[1][t] = __builtin_amdgcn_wmma_f32_16x16x32_f16(
                                false, a1, false, b, (short)0, acc[1][t], false, false);
            }
        }
        stage_wait();                                  // next panel resident
        __syncthreads();
    }

#pragma unroll
    for (int m = 0; m < 2; ++m) {
        const int rBase = m0 + m*16 + ((lane & 16) ? 8 : 0);
#pragma unroll
        for (int t = 0; t < 4; ++t) {
            int col = nBase + t*16 + nCol;
            float bv = bias[col];                  // N == fo for these layers
#pragma unroll
            for (int r = 0; r < 8; ++r)
                Y[(size_t)(rBase + r) * N + col] = acc[m][t][r] + bv;
        }
    }
}

// ---------------------------------------------------------------------------
// Tail GEMM (decoder 256->3, N padded to 16): 2 M-tiles per wave, B from L2.
// ---------------------------------------------------------------------------
__global__ __launch_bounds__(256)
void gemm_tail_k(const _Float16* __restrict__ A, const _Float16* __restrict__ Wt,
                 const float* __restrict__ bias, float* __restrict__ Y, int K)
{
    const int lane = threadIdx.x & 31;
    const int wv   = threadIdx.x >> 5;
    long w = (long)blockIdx.x * 8 + wv;
    if (w >= NPTS / 32) return;
    const int m0   = (int)w * 32;
    const int aSel = (lane & 16) ? 8 : 0;
    const int bOff = (lane & 16) ? 8 : 0;
    const int nCol = lane & 15;
    const _Float16* aPtr0 = A + (size_t)(m0 + nCol) * K + aSel;
    const _Float16* aPtr1 = aPtr0 + (size_t)16 * K;
    const _Float16* bPtr  = Wt + (size_t)nCol * K + bOff;

    v8f zero = {0.f,0.f,0.f,0.f,0.f,0.f,0.f,0.f};
    v8f acc0 = zero, acc1 = zero;
    for (int k0 = 0; k0 < K; k0 += 32) {
        v16h a0 = ldA(aPtr0 + k0);
        v16h a1 = ldA(aPtr1 + k0);
        v16h b  = ldB(bPtr + k0);
        acc0 = __builtin_amdgcn_wmma_f32_16x16x32_f16(false, a0, false, b, (short)0, acc0, false, false);
        acc1 = __builtin_amdgcn_wmma_f32_16x16x32_f16(false, a1, false, b, (short)0, acc1, false, false);
    }
    float bv = (nCol < 3) ? bias[nCol] : 0.0f;
    const int rb0 = m0 + ((lane & 16) ? 8 : 0);
#pragma unroll
    for (int r = 0; r < 8; ++r) {
        Y[(size_t)(rb0 + r)      * 16 + nCol] = acc0[r] + bv;
        Y[(size_t)(rb0 + 16 + r) * 16 + nCol] = acc1[r] + bv;
    }
}

// ---------------------------------------------------------------------------
// Column statistics (two-stage)
// ---------------------------------------------------------------------------
__global__ __launch_bounds__(256)
void col_partial_k(const float* __restrict__ Y, int N, int rowsPerChunk,
                   float* __restrict__ partial)
{
    int c = blockIdx.x * 256 + threadIdx.x;
    if (c >= N) return;
    int r0 = blockIdx.y * rowsPerChunk;
    float s = 0.f, s2 = 0.f;
    for (int r = r0; r < r0 + rowsPerChunk; ++r) {
        float v = Y[(size_t)r * N + c];
        s += v; s2 += v * v;
    }
    partial[((size_t)blockIdx.y * N + c) * 2 + 0] = s;
    partial[((size_t)blockIdx.y * N + c) * 2 + 1] = s2;
}

__global__ __launch_bounds__(256)
void col_finalize_k(const float* __restrict__ partial, int chunks, int N,
                    float invM, float* __restrict__ mean, float* __restrict__ rstd)
{
    int c = blockIdx.x * 256 + threadIdx.x;
    if (c >= N) return;
    float s = 0.f, s2 = 0.f;
    for (int ch = 0; ch < chunks; ++ch) {
        s  += partial[((size_t)ch * N + c) * 2 + 0];
        s2 += partial[((size_t)ch * N + c) * 2 + 1];
    }
    float m = s * invM;
    float var = s2 * invM - m * m;
    mean[c] = m;
    rstd[c] = rsqrtf(var + 1e-5f);
}

// ---------------------------------------------------------------------------
// BN + ReLU -> next layer's f16 A-matrix
// ---------------------------------------------------------------------------
__global__ __launch_bounds__(256)
void bn_relu_k(const float* __restrict__ Y, const float* __restrict__ mean,
               const float* __restrict__ rstd, const float* __restrict__ g,
               const float* __restrict__ t, _Float16* __restrict__ act,
               long total, int N)
{
    long idx = (long)blockIdx.x * 256 + threadIdx.x;
    if (idx >= total) return;
    int c = (int)(idx % N);
    float h = g[c] * (Y[idx] - mean[c]) * rstd[c] + t[c];
    act[idx] = (_Float16)(h > 0.0f ? h : 0.0f);
}

// ---------------------------------------------------------------------------
__global__ __launch_bounds__(256)
void tanh_pred_k(const float* __restrict__ Y16, float* __restrict__ pred)
{
    int idx = blockIdx.x * 256 + threadIdx.x;
    if (idx >= (int)(NPTS * 3)) return;
    int r = idx / 3, c = idx % 3;
    pred[idx] = tanhf(Y16[(size_t)r * 16 + c]);
}

// ---------------------------------------------------------------------------
// Per-batch NN search + weighted loss (targets staged in 12 KB LDS)
// ---------------------------------------------------------------------------
__global__ __launch_bounds__(256)
void nn_loss_k(const float* __restrict__ pin, const float* __restrict__ clean,
               const float* __restrict__ seeds, const float* __restrict__ stdv,
               const float* __restrict__ noise, float nscale,
               const float* __restrict__ pred, const float* __restrict__ sw,
               float* __restrict__ out)
{
    __shared__ float tgt[PPB * 3];
    __shared__ float red[256];
    int b = blockIdx.x, tid = threadIdx.x;
    float sb = (noise != nullptr) ? stdv[b] * nscale : 0.0f;
    float sx = seeds[b*3+0], sy = seeds[b*3+1], sz = seeds[b*3+2];
    for (int j = tid; j < PPB; j += 256) {
        int base = (b*PPB + j) * 3;
        float tx = clean[base+0] - sx, ty = clean[base+1] - sy, tz = clean[base+2] - sz;
        if (noise) { tx += noise[base+0]*sb; ty += noise[base+1]*sb; tz += noise[base+2]*sb; }
        tgt[j*3+0] = tx; tgt[j*3+1] = ty; tgt[j*3+2] = tz;
    }
    __syncthreads();
    float acc = 0.0f;
    for (int q = tid; q < PPB; q += 256) {
        int base = (b*PPB + q) * 3;
        float qx = pin[base+0], qy = pin[base+1], qz = pin[base+2];
        float best = 3.4e38f; int bi = 0;
        for (int r = 0; r < PPB; ++r) {
            float dx = qx - tgt[r*3+0], dy = qy - tgt[r*3+1], dz = qz - tgt[r*3+2];
            float d2 = dx*dx + dy*dy + dz*dz;
            if (d2 < best) { best = d2; bi = r; }
        }
        float cx = tgt[bi*3+0] - qx, cy = tgt[bi*3+1] - qy, cz = tgt[bi*3+2] - qz;
        float dx = pred[base+0] - cx, dy = pred[base+1] - cy, dz = pred[base+2] - cz;
        acc += sw[b*PPB + q] * (dx*dx + dy*dy + dz*dz);
    }
    red[tid] = acc; __syncthreads();
    for (int s = 128; s > 0; s >>= 1) { if (tid < s) red[tid] += red[tid+s]; __syncthreads(); }
    if (tid == 0) atomicAdd(out, red[0] * (1.0f / (float)BATCH));
}

// ---------------------------------------------------------------------------
extern "C" void kernel_launch(void* const* d_in, const int* in_sizes, int n_in,
                              void* d_out, int out_size, void* d_ws, size_t ws_size,
                              hipStream_t stream)
{
    (void)in_sizes; (void)n_in; (void)out_size; (void)ws_size;
    static const int LFI[8] = {3,64,128,256,512,1024,512,256};
    static const int LFO[8] = {64,128,256,512,1024,512,256,3};
    static const int LK [8] = {32,64,128,256,512,1024,512,256};
    static const int LN [8] = {64,128,256,512,1024,512,256,16};
    static const int WIX[8] = {5,9,13,17,21,25,29,33};
    static const int BIX[8] = {6,10,14,18,22,26,30,34};
    static const int GIX[8] = {7,11,15,19,23,27,31,-1};
    static const int HIX[8] = {8,12,16,20,24,28,32,-1};

    long wtOff[8]; long perIterHalves = 0;
    for (int l = 0; l < 8; ++l) { wtOff[l] = perIterHalves; perIterHalves += (long)LN[l]*LK[l]; }

    auto al = [](size_t x){ return (x + 255) & ~(size_t)255; };
    char* ws = (char*)d_ws;
    size_t off = 0;
    _Float16* wt   = (_Float16*)(ws + off); off = al(off + (size_t)perIterHalves * 4 * sizeof(_Float16));
    _Float16* act  = (_Float16*)(ws + off); off = al(off + (size_t)NPTS * 1024 * sizeof(_Float16));
    float*    Yb   = (float*)   (ws + off); off = al(off + (size_t)NPTS * 1024 * sizeof(float));
    float*    pin  = (float*)   (ws + off); off = al(off + (size_t)NPTS * 3 * sizeof(float));
    float*    pred = (float*)   (ws + off); off = al(off + (size_t)NPTS * 3 * sizeof(float));
    float*    sw   = (float*)   (ws + off); off = al(off + (size_t)BATCH * PPB * sizeof(float));
    float*    mean = (float*)   (ws + off); off = al(off + 1024 * sizeof(float));
    float*    rstd = (float*)   (ws + off); off = al(off + 1024 * sizeof(float));
    const int CHUNKS = 64, ROWS_PER_CHUNK = (int)(NPTS / 64);
    float*    part = (float*)   (ws + off); off = al(off + (size_t)CHUNKS * 1024 * 2 * sizeof(float));

    const float* noisy = (const float*)d_in[0];
    const float* clean = (const float*)d_in[1];
    const float* seeds = (const float*)d_in[2];
    const float* pstd  = (const float*)d_in[3];
    const float* noise = (const float*)d_in[4];
    float* out = (float*)d_out;

    zero_out_k<<<1, 1, 0, stream>>>(out);

    for (int l = 0; l < 8; ++l) {
        long tot = (long)LN[l] * LK[l];
        dim3 g((unsigned)((tot + 255) / 256), 1, 4);
        prep_wt_k<<<g, 256, 0, stream>>>((const float*)d_in[WIX[l]], wt + wtOff[l],
                                         LFI[l], LFO[l], LK[l], LN[l], perIterHalves);
    }

    seed_sw_k<<<BATCH, 256, 0, stream>>>(noisy, seeds, sw);

    for (int it = 0; it < 4; ++it) {
        prep_input_k<<<(unsigned)((NPTS + 255) / 256), 256, 0, stream>>>(
            noisy, seeds, pred, pin, act, it);

        const _Float16* wtIter = wt + (long)it * perIterHalves;
        for (int l = 0; l < 8; ++l) {
            int K = LK[l], N = LN[l], fo = LFO[l];
            const float* bias = (const float*)d_in[BIX[l]] + (long)it * fo;

            if (l < 7) {
                int Nq = N >> 6;                       // N/64 panels
                unsigned blocks = (unsigned)(125 * Nq);  // 32000/256 row groups
                gemm_lds_k<<<blocks, 256, 0, stream>>>(act, wtIter + wtOff[l],
                                                       bias, Yb, K, N, Nq);
                dim3 gp((unsigned)((N + 255) / 256), CHUNKS);
                col_partial_k<<<gp, 256, 0, stream>>>(Yb, N, ROWS_PER_CHUNK, part);
                col_finalize_k<<<(unsigned)((N + 255) / 256), 256, 0, stream>>>(
                    part, CHUNKS, N, 1.0f / (float)NPTS, mean, rstd);
                const float* g = (const float*)d_in[GIX[l]] + (long)it * fo;
                const float* t = (const float*)d_in[HIX[l]] + (long)it * fo;
                long total = NPTS * (long)N;
                bn_relu_k<<<(unsigned)((total + 255) / 256), 256, 0, stream>>>(
                    Yb, mean, rstd, g, t, act, total, N);
            } else {
                gemm_tail_k<<<125, 256, 0, stream>>>(act, wtIter + wtOff[l],
                                                     bias, Yb, K);
                tanh_pred_k<<<(unsigned)((NPTS * 3 + 255) / 256), 256, 0, stream>>>(Yb, pred);
            }
        }

        const float* noise_i = (it < 3) ? (noise + (long)it * BATCH * PPB * 3) : nullptr;
        float nscale = (it == 0) ? 0.25f : (it == 1) ? 0.0625f : 0.015625f;
        nn_loss_k<<<BATCH, 256, 0, stream>>>(pin, clean, seeds, pstd,
                                             noise_i, nscale, pred, sw, out);
    }
}
